// InterestTransformerEncoder_64982855188795
// MI455X (gfx1250) — matrix operand order repaired and checked by reading, not compiled, not run on hardware
//
#include <hip/hip_runtime.h>

typedef __attribute__((ext_vector_type(16))) _Float16 v16h;
typedef __attribute__((ext_vector_type(8)))  _Float16 v8h;
typedef __attribute__((ext_vector_type(8)))  float    v8f;

#define B_    1024
#define L_    200
#define D_    64
#define LP    224   // L padded to 14 tiles of 16 (7 K-chunks of 32)
#define NT    14    // LP/16 row tiles
#define NWAVE 8

static __device__ __forceinline__ v8f wmma_f16(v16h a, v16h b, v8f c) {
  return __builtin_amdgcn_wmma_f32_16x16x32_f16(false, a, false, b, (short)0, c,
                                                false, false);
}

// Combine two aligned 16-byte LDS runs into one 16-half fragment.
static __device__ __forceinline__ v16h ldpair(const _Float16* p0,
                                              const _Float16* p1) {
  v8h a = *(const v8h*)p0;
  v8h b = *(const v8h*)p1;
  return __builtin_shufflevector(a, b, 0, 1, 2, 3, 4, 5, 6, 7,
                                 8, 9, 10, 11, 12, 13, 14, 15);
}

// A-fragment from a K-contiguous row: lane m=lh, K runs {kb+hi*8..+7} and
// {kb+16+hi*8..+7}  (ISA 16-bit A layout).
static __device__ __forceinline__ v16h loadA_row(const _Float16* row, int kb,
                                                 int hi) {
  return ldpair(row + kb + hi * 8, row + kb + 16 + hi * 8);
}

// B-fragment: lane n=lh reads K runs {kb+hi*8..+7} and {kb+16+hi*8..+7}.
static __device__ __forceinline__ v16h loadB_row(const _Float16* row, int kb,
                                                 int hi) {
  const _Float16* p = row + kb + hi * 8;
  return ldpair(p, p + 16);
}

__global__ __launch_bounds__(256, 2)
void interest_encoder_kernel(const int* __restrict__ seq,
                             const float* __restrict__ hist,
                             const float* __restrict__ W,
                             float* __restrict__ out) {
  __shared__ alignas(16) _Float16 sHn[LP][D_];         // normalized h, row-major
  __shared__ alignas(16) _Float16 sHnT[D_][LP];        // normalized h, transposed
  __shared__ alignas(16) _Float16 sWT[D_][D_];         // W transposed: sWT[e][d]
  __shared__ float    sCmask[LP];                      // mask_j
  __shared__ float    sCscale[LP];                     // mask_j * sqrt(ss+0.01)
  __shared__ alignas(16) _Float16 sScr[NWAVE][16][D_]; // per-wave staging

  const int b    = blockIdx.x;
  const int tid  = threadIdx.x;
  const int wave = tid >> 5;
  const int lane = tid & 31;
  const int lh   = lane & 15;
  const int hi   = lane >> 4;

  // ---- Phase 0a: W -> f16 transposed ----
  for (int idx = tid; idx < D_ * D_; idx += 256) {
    const int e = idx >> 6, d = idx & 63;
    sWT[e][d] = (_Float16)W[d * D_ + e];
  }

  // ---- Phase 0b: per column j: norms, masks, both Hn orientations ----
  if (tid < LP) {
    const int j = tid;
    if (j < L_) {
      const float4* hj4 = (const float4*)(hist + ((size_t)b * L_ + j) * D_);
      float4 hv[16];
      float ss = 0.f;
#pragma unroll
      for (int q = 0; q < 16; ++q) {
        hv[q] = hj4[q];
        ss += hv[q].x * hv[q].x + hv[q].y * hv[q].y +
              hv[q].z * hv[q].z + hv[q].w * hv[q].w;
      }
      const float inv = rsqrtf(ss + 0.01f);
      const float nrm = sqrtf(ss + 0.01f);
      const float m   = (seq[(size_t)b * L_ + j] != 0) ? 1.f : 0.f;
      sCmask[j]  = m;
      sCscale[j] = m * nrm;
#pragma unroll
      for (int q = 0; q < 8; ++q) {
        v8h r;
        r[0] = (_Float16)(hv[2 * q].x * inv);
        r[1] = (_Float16)(hv[2 * q].y * inv);
        r[2] = (_Float16)(hv[2 * q].z * inv);
        r[3] = (_Float16)(hv[2 * q].w * inv);
        r[4] = (_Float16)(hv[2 * q + 1].x * inv);
        r[5] = (_Float16)(hv[2 * q + 1].y * inv);
        r[6] = (_Float16)(hv[2 * q + 1].z * inv);
        r[7] = (_Float16)(hv[2 * q + 1].w * inv);
        *(v8h*)&sHn[j][q * 8] = r;
#pragma unroll
        for (int t = 0; t < 8; ++t) sHnT[q * 8 + t][j] = r[t];
      }
    } else {
      sCmask[j] = 0.f; sCscale[j] = 0.f;
      v8h z = (v8h)((_Float16)0.f);
#pragma unroll
      for (int q = 0; q < 8; ++q) {
        *(v8h*)&sHn[j][q * 8] = z;
#pragma unroll
        for (int t = 0; t < 8; ++t) sHnT[q * 8 + t][j] = (_Float16)0.f;
      }
    }
  }
  __syncthreads();

  _Float16 (*scr)[D_] = sScr[wave];

  // ---- Phase 1: each wave owns row-tiles mt, mt+8 ----
  for (int mt = wave; mt < NT; mt += NWAVE) {
    const int i0 = mt * 16;

    // GEMM1 A fragments (rows i0..i0+15, K=d in 2 chunks of 32)
    const v16h a0 = loadA_row(sHn[i0 + lh], 0, hi);
    const v16h a1 = loadA_row(sHn[i0 + lh], 32, hi);

    float dsum[8];
#pragma unroll
    for (int r = 0; r < 8; ++r) dsum[r] = 0.f;
    v8f hacc[4];
#pragma unroll
    for (int nd = 0; nd < 4; ++nd) hacc[nd] = (v8f)(0.f);

    for (int kc = 0; kc < LP / 32; ++kc) {   // 7 chunks of 32 j-columns
      const int j0 = kc * 32;
#pragma unroll
      for (int sub = 0; sub < 2; ++sub) {
        const int n0 = j0 + sub * 16;
        // GEMM1 B: B[k=d][n=j] = sHn[n0+lh][k], K-contiguous
        v8f acc = (v8f)(0.f);
        acc = wmma_f16(a0, loadB_row(sHn[n0 + lh], 0, hi), acc);
        acc = wmma_f16(a1, loadB_row(sHn[n0 + lh], 32, hi), acc);
        // weights for this 16x16 score tile (C layout)
        const int   j  = n0 + lh;
        const float cm = sCmask[j];
        const float cs = sCscale[j];
#pragma unroll
        for (int r = 0; r < 8; ++r) {
          const float wv = __expf(acc[r]);
          dsum[r] += wv * cm;                                   // denom
          scr[r + hi * 8][sub * 16 + lh] = (_Float16)(wv * cs); // w * norm_j
        }
      }
      // GEMM2 partial: h_acc += w_chunk(16x32) x Hn(32x64)
      const v16h wa = loadA_row(scr[lh], 0, hi);
#pragma unroll
      for (int nd = 0; nd < 4; ++nd)
        hacc[nd] = wmma_f16(wa, loadB_row(sHnT[nd * 16 + lh], j0, hi), hacc[nd]);
    }

    // reduce denom across the 16-lane half that shares each row
#pragma unroll
    for (int r = 0; r < 8; ++r) {
      float v = dsum[r];
      v += __shfl_xor(v, 1);
      v += __shfl_xor(v, 2);
      v += __shfl_xor(v, 4);
      v += __shfl_xor(v, 8);
      dsum[r] = v;
    }

    // t = h_acc/denom + h (residual), staged f16 as GEMM3 A-operand
#pragma unroll
    for (int nd = 0; nd < 4; ++nd) {
#pragma unroll
      for (int r = 0; r < 8; ++r) {
        const int irow = i0 + r + hi * 8;
        const int d    = nd * 16 + lh;
        const float hv = (irow < L_) ? hist[((size_t)b * L_ + irow) * D_ + d]
                                     : 0.f;
        scr[r + hi * 8][d] = (_Float16)(hacc[nd][r] / dsum[r] + hv);
      }
    }

    // GEMM3: (16x64) x W(64x64)
    v8f o[4];
#pragma unroll
    for (int ne = 0; ne < 4; ++ne) o[ne] = (v8f)(0.f);
#pragma unroll
    for (int kd = 0; kd < 2; ++kd) {
      const v16h ta = loadA_row(scr[lh], kd * 32, hi);
#pragma unroll
      for (int ne = 0; ne < 4; ++ne)
        o[ne] = wmma_f16(ta, loadB_row(sWT[ne * 16 + lh], kd * 32, hi), o[ne]);
    }

    // relu + query mask + store
#pragma unroll
    for (int r = 0; r < 8; ++r) {
      const int irow = i0 + r + hi * 8;
      if (irow < L_) {
        const float mi = (seq[(size_t)b * L_ + irow] != 0) ? 1.f : 0.f;
#pragma unroll
        for (int ne = 0; ne < 4; ++ne) {
          float v = o[ne][r];
          v = v > 0.f ? v : 0.f;
          out[((size_t)b * L_ + irow) * D_ + ne * 16 + lh] = v * mi;
        }
      }
    }
  }
}

extern "C" void kernel_launch(void* const* d_in, const int* in_sizes, int n_in,
                              void* d_out, int out_size, void* d_ws, size_t ws_size,
                              hipStream_t stream) {
  (void)in_sizes; (void)n_in; (void)d_ws; (void)ws_size; (void)out_size;
  const int*   seq  = (const int*)d_in[0];
  const float* hist = (const float*)d_in[1];
  const float* W    = (const float*)d_in[2];
  float*       out  = (float*)d_out;
  interest_encoder_kernel<<<dim3(B_), dim3(256), 0, stream>>>(seq, hist, W, out);
}
